// w2v_model_71004399338132
// MI455X (gfx1250) — matrix operand order repaired and checked by reading, not compiled, not run on hardware
//
#include <hip/hip_runtime.h>

// ---------------------------------------------------------------------------
// MI455X (gfx1250) fused word2vec-attention model.
//
// Roofline: projections dominate (~146 GFLOP bf16-WMMA); HBM traffic only
// ~180 MB if K/V never touch HBM -> fuse K/V projection with attention,
// staging K/V tiles in LDS (bf16). Projections and scores via v_wmma_f32_16x16x32_bf16; A@V on VALU.
// Embedding gathers use GLOBAL_LOAD_ASYNC_TO_LDS_B128 (ASYNCcnt-tracked,
// L2 -> LDS without VGPR round trip), then a bf16 conversion pass in LDS.
// ---------------------------------------------------------------------------

typedef __attribute__((ext_vector_type(16))) __bf16 v16bf;
typedef __attribute__((ext_vector_type(2)))  __bf16 bf16x2;
typedef __attribute__((ext_vector_type(8)))  float  v8f;

#define BATCH  4096
#define SEQ    16
#define EMBED  512
#define HIDDEN 1024
#define NHEAD  16
#define DHEAD  64

// LDS byte offset of a generic pointer into shared memory: flat->LDS aperture
// mapping truncates to addr[31:0] (ISA 10.2), which is the wave-relative LDS
// address the async-load VDST operand expects.
__device__ __forceinline__ unsigned lds_off(const void* p) {
    return (unsigned)(unsigned long long)p;
}

// Async copy of 16 bytes global -> LDS (tracked with ASYNCcnt).
__device__ __forceinline__ void async_g2lds_b128(unsigned ldsaddr, const void* gaddr) {
    asm volatile("global_load_async_to_lds_b128 %0, %1, off"
                 :: "v"(ldsaddr), "v"(gaddr) : "memory");
}
__device__ __forceinline__ void wait_async0() {
    asm volatile("s_wait_asynccnt 0" ::: "memory");
}

// ---------------------------------------------------------------------------
// Kernel 0: zero context-vector output half (atomic accumulator) and convert
// WQ/WK/WV f32 -> bf16 into workspace (halves weight re-stream traffic).
// ---------------------------------------------------------------------------
__global__ void k_prep(const float* __restrict__ WQ, const float* __restrict__ WK,
                       const float* __restrict__ WV,
                       __bf16* __restrict__ wqb, __bf16* __restrict__ wkb,
                       __bf16* __restrict__ wvb, float* __restrict__ ctx_out) {
    const long NOUT = (long)BATCH * HIDDEN;          // 4194304
    const long NW   = (long)HIDDEN * EMBED;          // 524288
    long stride = (long)gridDim.x * blockDim.x;
    for (long i = (long)blockIdx.x * blockDim.x + threadIdx.x; i < NOUT; i += stride) {
        ctx_out[i] = 0.0f;
        if (i < NW) {
            wqb[i] = (__bf16)WQ[i];
            wkb[i] = (__bf16)WK[i];
            wvb[i] = (__bf16)WV[i];
        }
    }
}

// A-fragment K index for 16-bit A 16x32 (ISA 7.12.2): VGPR j<4 -> K=2j(+8*half),
// VGPR j>=4 -> K=16+2(j-4)(+8*half); elements 2j,2j+1 hold K, K+1.
__device__ __forceinline__ int a_koff(int j, int half) {
    return ((j < 4) ? (2 * j) : (16 + 2 * (j - 4))) + 8 * half;
}

// ---------------------------------------------------------------------------
// Kernel 1: Q = target @ WQ^T + bQ  (bf16 -> workspace)
//           target_vector = target @ WV^T + bV (f32 -> d_out first half)
// Grid: (256 batch tiles of 16, 8); 8 waves/block, wave owns one 16-col tile.
// ---------------------------------------------------------------------------
__global__ void k_proj_q_tv(const int* __restrict__ t, const float* __restrict__ emb,
                            const __bf16* __restrict__ wqb, const __bf16* __restrict__ wvb,
                            const float* __restrict__ bQ, const float* __restrict__ bV,
                            __bf16* __restrict__ qout, float* __restrict__ tvout) {
    __shared__ float  sStage[16 * EMBED];            // 32 KB: async-gathered f32 rows
    __shared__ __bf16 sA[16 * EMBED];                // 16 KB: bf16 A operand

    const int b0 = blockIdx.x * 16;

    // async gather: 2048 b128 transfers, 8 per thread (per-lane addresses)
#pragma unroll
    for (int v = 0; v < 8; ++v) {
        int i   = ((int)threadIdx.x + v * 256) * 4;  // f32 element index, 16B aligned
        int r   = i >> 9, col = i & 511;
        const float* g = &emb[(long)t[b0 + r] * EMBED + col];
        async_g2lds_b128(lds_off(&sStage[i]), g);
    }
    wait_async0();
    __syncthreads();
    for (int i = threadIdx.x * 4; i < 16 * EMBED; i += 1024) {
        sA[i + 0] = (__bf16)sStage[i + 0];
        sA[i + 1] = (__bf16)sStage[i + 1];
        sA[i + 2] = (__bf16)sStage[i + 2];
        sA[i + 3] = (__bf16)sStage[i + 3];
    }
    __syncthreads();

    const int lane = threadIdx.x & 31;
    const int w    = threadIdx.x >> 5;
    const int m    = lane & 15;
    const int half = lane >> 4;
    const int n0   = (blockIdx.y * 8 + w) * 16;

    v8f cq = {};
    v8f cv = {};
    for (int kk = 0; kk < 16; ++kk) {
        const int k0 = kk * 32;
        v16bf a, bq, bv;
#pragma unroll
        for (int j = 0; j < 8; ++j) {
            // A: row m of gathered targets
            bf16x2 pa = *(const bf16x2*)&sA[m * EMBED + k0 + a_koff(j, half)];
            a[2 * j] = pa[0]; a[2 * j + 1] = pa[1];
            // B(k,n) = W[n*512 + k]; K in same table order as A
            int kb = k0 + a_koff(j, half);
            long wo = (long)(n0 + m) * EMBED + kb;
            bf16x2 pq = *(const bf16x2*)&wqb[wo];
            bq[2 * j] = pq[0]; bq[2 * j + 1] = pq[1];
            bf16x2 pv = *(const bf16x2*)&wvb[wo];
            bv[2 * j] = pv[0]; bv[2 * j + 1] = pv[1];
        }
        cq = __builtin_amdgcn_wmma_f32_16x16x32_bf16(false, a, false, bq, (short)0, cq, false, false);
        cv = __builtin_amdgcn_wmma_f32_16x16x32_bf16(false, a, false, bv, (short)0, cv, false, false);
    }

    const int n = n0 + (lane & 15);
#pragma unroll
    for (int r_ = 0; r_ < 8; ++r_) {
        long row = b0 + r_ + 8 * half;
        qout[row * HIDDEN + n]  = (__bf16)(cq[r_] + bQ[n]);
        tvout[row * HIDDEN + n] = cv[r_] + bV[n];
    }
}

// ---------------------------------------------------------------------------
// Kernel 2: fused K/V projection + head-x-head attention for one (16-batch, s).
// Phase 1: 8 waves compute 128 WMMA tiles (64 K + 64 V) into LDS (bf16 + bias).
// Phase 2: each wave handles 2 batch rows: scores = Qh @ Kh^T (2x WMMA K=32),
//          softmax across the 16-lane N axis, VALU A@V, atomic add into out.
// ---------------------------------------------------------------------------
__global__ void k_attn(const int* __restrict__ c, const float* __restrict__ emb,
                       const __bf16* __restrict__ wkb, const __bf16* __restrict__ wvb,
                       const float* __restrict__ bK, const float* __restrict__ bV,
                       const __bf16* __restrict__ qb, float* __restrict__ ctx_out) {
    extern __shared__ __align__(16) char smem[];
    __bf16* sK  = (__bf16*)smem;                // 16 x 1024 bf16  (32 KB)
    __bf16* sV  = sK + 16 * HIDDEN;             // 16 x 1024 bf16  (32 KB)
    __bf16* sC  = sV + 16 * HIDDEN;             // 16 x 512  bf16  (16 KB)
    float*  sAw = (float*)(sC + 16 * EMBED);    // 8 waves x 16x16 f32 (8 KB)
    float*  sCf = sAw + 8 * 256;                // 16 x 512  f32   (32 KB staging)

    const int b0 = blockIdx.x * 16;
    const int s  = blockIdx.y;

    // async gather of 16 context rows (this s) into f32 staging
#pragma unroll
    for (int v = 0; v < 8; ++v) {
        int i   = ((int)threadIdx.x + v * 256) * 4;
        int r   = i >> 9, col = i & 511;
        int tok = c[(b0 + r) * SEQ + s];
        async_g2lds_b128(lds_off(&sCf[i]), &emb[(long)tok * EMBED + col]);
    }
    wait_async0();
    __syncthreads();
    for (int i = threadIdx.x * 4; i < 16 * EMBED; i += 1024) {
        sC[i + 0] = (__bf16)sCf[i + 0];
        sC[i + 1] = (__bf16)sCf[i + 1];
        sC[i + 2] = (__bf16)sCf[i + 2];
        sC[i + 3] = (__bf16)sCf[i + 3];
    }
    __syncthreads();

    const int lane = threadIdx.x & 31;
    const int w    = threadIdx.x >> 5;
    const int m    = lane & 15;
    const int half = lane >> 4;

    // ---- Phase 1: K_s and V_s projections into LDS ----
    for (int tile = w; tile < 128; tile += 8) {
        const bool isV = tile >= 64;
        const int  nt  = isV ? tile - 64 : tile;
        const __bf16* wb   = isV ? wvb : wkb;
        const float*  bias = isV ? bV  : bK;
        __bf16*       dst  = isV ? sV  : sK;
        const int n0 = nt * 16;

        // prefetch next tile's weight lines while this tile's k-loop runs
        if (tile + 8 < 128) {
            const __bf16* nwb = (tile + 8 >= 64) ? wvb : wkb;
            int nn0 = ((tile + 8 >= 64) ? (tile + 8 - 64) : (tile + 8)) * 16;
            __builtin_prefetch(&nwb[(long)(nn0 + m) * EMBED + 32 * half], 0, 1);
            __builtin_prefetch(&nwb[(long)(nn0 + m) * EMBED + 256 + 32 * half], 0, 1);
        }

        v8f acc = {};
        for (int kk = 0; kk < 16; ++kk) {
            const int k0 = kk * 32;
            v16bf a, b;
#pragma unroll
            for (int j = 0; j < 8; ++j) {
                bf16x2 pa = *(const bf16x2*)&sC[m * EMBED + k0 + a_koff(j, half)];
                a[2 * j] = pa[0]; a[2 * j + 1] = pa[1];
                int kb = k0 + a_koff(j, half);
                bf16x2 pb = *(const bf16x2*)&wb[(long)(n0 + m) * EMBED + kb];
                b[2 * j] = pb[0]; b[2 * j + 1] = pb[1];
            }
            acc = __builtin_amdgcn_wmma_f32_16x16x32_bf16(false, a, false, b, (short)0, acc, false, false);
        }
        const int n = n0 + (lane & 15);
#pragma unroll
        for (int r_ = 0; r_ < 8; ++r_)
            dst[(r_ + 8 * half) * HIDDEN + n] = (__bf16)(acc[r_] + bias[n]);
    }
    __syncthreads();

    // ---- Phase 2: attention; wave w handles local batch rows 2w, 2w+1 ----
    for (int rr = 0; rr < 2; ++rr) {
        const int r = w * 2 + rr;
        const __bf16* qrow = qb + (long)(b0 + r) * HIDDEN;

        // scores(h,g) = sum_d Qh(h,d) * Kh(g,d); M=h, N=g, K=d=64 -> 2 WMMAs
        v8f sc = {};
        for (int k0 = 0; k0 <= 32; k0 += 32) {
            v16bf a, b;
#pragma unroll
            for (int j = 0; j < 8; ++j) {
                int ka = k0 + a_koff(j, half);
                bf16x2 pa = *(const bf16x2*)&qrow[m * DHEAD + ka];   // A(h=m, d=ka)
                a[2 * j] = pa[0]; a[2 * j + 1] = pa[1];
                int kb = k0 + a_koff(j, half);                       // B(d=kb, g=m)
                bf16x2 pb = *(const bf16x2*)&sK[r * HIDDEN + m * DHEAD + kb];
                b[2 * j] = pb[0]; b[2 * j + 1] = pb[1];
            }
            sc = __builtin_amdgcn_wmma_f32_16x16x32_bf16(false, a, false, b, (short)0, sc, false, false);
        }

        // softmax over g (N axis = across each 16-lane group), write A to LDS
#pragma unroll
        for (int r_ = 0; r_ < 8; ++r_) {
            float v = sc[r_] * 0.125f;                // 1/sqrt(64)
            float mx = v;
            for (int msk = 1; msk < 16; msk <<= 1)
                mx = fmaxf(mx, __shfl_xor(mx, msk, 32));
            float e = __expf(v - mx);
            float sum = e;
            for (int msk = 1; msk < 16; msk <<= 1)
                sum += __shfl_xor(sum, msk, 32);
            sAw[w * 256 + (r_ + 8 * half) * 16 + (lane & 15)] = e / sum;
        }
        __builtin_amdgcn_wave_barrier();  // LDS ops are in-order within a wave

        // tmp(h,d) = sum_g A(h,g) * Vh(g,d); accumulate over s via global atomics
        for (int o = lane; o < HIDDEN; o += 32) {
            const int h = o >> 6, d = o & 63;
            float acc = 0.0f;
#pragma unroll
            for (int g = 0; g < 16; ++g)
                acc += sAw[w * 256 + h * 16 + g] * (float)sV[r * HIDDEN + g * DHEAD + d];
            atomicAdd(&ctx_out[(long)(b0 + r) * HIDDEN + o], acc);
        }
    }
}

// ---------------------------------------------------------------------------
extern "C" void kernel_launch(void* const* d_in, const int* in_sizes, int n_in,
                              void* d_out, int out_size, void* d_ws, size_t ws_size,
                              hipStream_t stream) {
    const int*   t   = (const int*)d_in[0];
    const int*   c   = (const int*)d_in[1];
    const float* emb = (const float*)d_in[2];
    const float* WQ  = (const float*)d_in[3];
    const float* bQ  = (const float*)d_in[4];
    const float* WK  = (const float*)d_in[5];
    const float* bK  = (const float*)d_in[6];
    const float* WV  = (const float*)d_in[7];
    const float* bV  = (const float*)d_in[8];

    float* tv_out  = (float*)d_out;                          // [4096,1024]
    float* ctx_out = tv_out + (long)BATCH * HIDDEN;          // [4096,1024]

    // workspace: bf16 weights (3 x 512K) + bf16 Q (4096x1024) = 11 MB
    __bf16* wqb = (__bf16*)d_ws;
    __bf16* wkb = wqb + (long)HIDDEN * EMBED;
    __bf16* wvb = wkb + (long)HIDDEN * EMBED;
    __bf16* qb  = wvb + (long)HIDDEN * EMBED;

    k_prep<<<4096, 256, 0, stream>>>(WQ, WK, WV, wqb, wkb, wvb, ctx_out);

    k_proj_q_tv<<<dim3(BATCH / 16, 8), 256, 0, stream>>>(t, emb, wqb, wvb, bQ, bV, qb, tv_out);

    const size_t smem = (16 * HIDDEN + 16 * HIDDEN + 16 * EMBED) * sizeof(__bf16)
                      + 8 * 256 * sizeof(float)
                      + 16 * EMBED * sizeof(float);          // 120 KB dynamic LDS
    k_attn<<<dim3(BATCH / 16, SEQ), 256, smem, stream>>>(c, emb, wkb, wvb, bK, bV, qb, ctx_out);
}
